// MultiHeadTargetAttention_9088150798340
// MI455X (gfx1250) — matrix operand order repaired and checked by reading, not compiled, not run on hardware
//
#include <hip/hip_runtime.h>
#include <hip/hip_bf16.h>

#define EMB 128
#define HEADS 4
#define HEAD_DIM 32
#define BATCH 4096
#define SEQLEN 200
#define TILE 16
#define NTILES ((SEQLEN + TILE - 1) / TILE)   // 13
#define KV_STR 132        // f32 elements per row (padded, 4-bank skew)
#define SEQH_STR 136      // bf16 elements per row (padded, multiple of 8)
#define SEQH_USTR (SEQH_STR / 2)              // 68 uints per row
#define LN_EPS 1e-5f
#define INV_SQRT_D 0.17677669529663687f       // 1/sqrt(32)

typedef __bf16 v16bf __attribute__((ext_vector_type(16)));
typedef float  v8f   __attribute__((ext_vector_type(8)));
typedef float  v4f   __attribute__((ext_vector_type(4)));

__device__ __forceinline__ unsigned short bf16_bits(__bf16 h) {
    union { __bf16 b; unsigned short u; } cv; cv.b = h; return cv.u;
}

// split x into bf16 hi + bf16 lo (round-to-nearest each step)
__device__ __forceinline__ void bf16_split(float x, unsigned short& h, unsigned short& l) {
    __bf16 hb = (__bf16)x;
    float r = x - (float)hb;
    __bf16 lb = (__bf16)r;
    h = bf16_bits(hb);
    l = bf16_bits(lb);
}

union ABFrag { v16bf bf; uint4 u2[2]; };

__global__ void __launch_bounds__(256)
mhta_kernel(const float* __restrict__ target,
            const float* __restrict__ seq,
            const unsigned char* __restrict__ mask,
            const float* __restrict__ Wq, const float* __restrict__ bq,
            const float* __restrict__ Wk, const float* __restrict__ bk,
            const float* __restrict__ Wv, const float* __restrict__ bv,
            const float* __restrict__ Wo, const float* __restrict__ bo,
            const float* __restrict__ gamma, const float* __restrict__ beta,
            float* __restrict__ out)
{
    __shared__ unsigned int lds_shi[TILE * SEQH_USTR];  // seq tile, bf16 hi, packed pairs
    __shared__ unsigned int lds_slo[TILE * SEQH_USTR];  // seq tile, bf16 lo
    __shared__ float lds_k[TILE * KV_STR];              // K tile (f32, bias added)
    __shared__ float lds_v[TILE * KV_STR];              // V tile (f32, bias added)
    __shared__ float q_s[EMB];
    __shared__ float t_s[EMB];
    __shared__ float score_s[TILE * HEADS];
    __shared__ float acc_s[EMB];
    __shared__ float o_s[EMB];

    const int tid  = threadIdx.x;
    const int b    = blockIdx.x;
    const int wave = tid >> 5;      // 0..7 -> N-tile (16 output features each)
    const int lane = tid & 31;
    const int n    = lane & 15;
    const int khi  = lane >> 4;     // lane-group for A/B K-striping
    const int feat = wave * 16 + n; // this lane's output-feature column

    // ---- stage target row; compute q in f32 ----
    if (tid < EMB) t_s[tid] = target[(size_t)b * EMB + tid];
    __syncthreads();
    if (tid < EMB) {
        float acc = bq[tid];
        const float* wrow = Wq + (size_t)tid * EMB;
        #pragma unroll 8
        for (int k = 0; k < EMB; ++k) acc += t_s[k] * wrow[k];
        q_s[tid] = acc;
    }

    // ---- build register-resident bf16 hi/lo B fragments for Wk, Wv ----
    // B layout (32x16 bf16): lane (16*khi + n) holds K = 32*kc + 16*khi + (0..15)
    v16bf bkhi[4], bklo[4], bvhi[4], bvlo[4];
    {
        const float* kr = Wk + (size_t)feat * EMB + khi * 16;
        const float* vr = Wv + (size_t)feat * EMB + khi * 16;
        #pragma unroll
        for (int kc = 0; kc < 4; ++kc) {
            #pragma unroll
            for (int e = 0; e < 16; ++e) {
                unsigned short h, l;
                bf16_split(kr[kc * 32 + e], h, l);
                bkhi[kc][e] = __builtin_bit_cast(__bf16, h);
                bklo[kc][e] = __builtin_bit_cast(__bf16, l);
                bf16_split(vr[kc * 32 + e], h, l);
                bvhi[kc][e] = __builtin_bit_cast(__bf16, h);
                bvlo[kc][e] = __builtin_bit_cast(__bf16, l);
            }
        }
    }
    const float bkb = bk[feat];
    const float bvb = bv[feat];

    float attnAcc = 0.0f;                 // attention output, feature = tid (tid < 128)
    const int myHead = (tid & 127) >> 5;  // feature tid belongs to head tid/32

    for (int tile = 0; tile < NTILES; ++tile) {
        const int rowBase = tile * TILE;
        __syncthreads();  // previous tile's consumers done before overwriting LDS

        // ---- stage seq tile: split to bf16 hi/lo once, shared by all waves ----
        {
            const int r  = tid >> 4;          // 0..15
            const int c  = (tid & 15) * 8;    // 0,8,...,120
            const int gr = rowBase + r;
            unsigned int hp[4] = {0u, 0u, 0u, 0u};
            unsigned int lp[4] = {0u, 0u, 0u, 0u};
            if (gr < SEQLEN) {
                const float* src = seq + ((size_t)b * SEQLEN + gr) * EMB + c;
                v4f a0 = *(const v4f*)(src);
                v4f a1 = *(const v4f*)(src + 4);
                float x[8] = {a0[0], a0[1], a0[2], a0[3], a1[0], a1[1], a1[2], a1[3]};
                #pragma unroll
                for (int p = 0; p < 4; ++p) {
                    unsigned short h0, l0, h1, l1;
                    bf16_split(x[2 * p],     h0, l0);
                    bf16_split(x[2 * p + 1], h1, l1);
                    hp[p] = (unsigned int)h0 | ((unsigned int)h1 << 16);
                    lp[p] = (unsigned int)l0 | ((unsigned int)l1 << 16);
                }
            }
            unsigned int base = r * SEQH_USTR + (c >> 1);
            *(uint4*)&lds_shi[base] = make_uint4(hp[0], hp[1], hp[2], hp[3]);
            *(uint4*)&lds_slo[base] = make_uint4(lp[0], lp[1], lp[2], lp[3]);
        }
        __syncthreads();

        // ---- WMMA: K-tile and V-tile (bf16x3) for this wave's 16 features ----
        v8f acck = {};
        v8f accv = {};
        {
            const int m = n;  // A-matrix row index = lane & 15
            // A layout (16x32 bf16): lane holds K = 32*kc + 16*khi + (0..15)
            const unsigned int aoff = m * SEQH_USTR + khi * 8;
            #pragma unroll
            for (int kc = 0; kc < 4; ++kc) {
                const unsigned int kb = kc * 16;  // in uints: 32 bf16 elems
                ABFrag ahi, alo;
                ahi.u2[0] = *(const uint4*)&lds_shi[aoff + kb];
                ahi.u2[1] = *(const uint4*)&lds_shi[aoff + kb + 4];
                alo.u2[0] = *(const uint4*)&lds_slo[aoff + kb];
                alo.u2[1] = *(const uint4*)&lds_slo[aoff + kb + 4];
                acck = __builtin_amdgcn_wmma_f32_16x16x32_bf16(false, ahi.bf, false, bkhi[kc], (short)0, acck, false, false);
                acck = __builtin_amdgcn_wmma_f32_16x16x32_bf16(false, ahi.bf, false, bklo[kc], (short)0, acck, false, false);
                acck = __builtin_amdgcn_wmma_f32_16x16x32_bf16(false, alo.bf, false, bkhi[kc], (short)0, acck, false, false);
                accv = __builtin_amdgcn_wmma_f32_16x16x32_bf16(false, ahi.bf, false, bvhi[kc], (short)0, accv, false, false);
                accv = __builtin_amdgcn_wmma_f32_16x16x32_bf16(false, ahi.bf, false, bvlo[kc], (short)0, accv, false, false);
                accv = __builtin_amdgcn_wmma_f32_16x16x32_bf16(false, alo.bf, false, bvhi[kc], (short)0, accv, false, false);
            }
        }
        // C/D layout: VGPR j, lanes0-15 -> (row j, col n); lanes16-31 -> (row 8+j, col n)
        #pragma unroll
        for (int j = 0; j < 8; ++j) {
            const int row = j + 8 * khi;
            lds_k[row * KV_STR + feat] = acck[j] + bkb;
            lds_v[row * KV_STR + feat] = accv[j] + bvb;
        }
        __syncthreads();

        // ---- scores: threads 0..63 -> (l, h), f32 ----
        if (tid < TILE * HEADS) {
            const int l  = tid & 15;
            const int h  = tid >> 4;
            const int gl = rowBase + l;
            float s = 0.0f;
            if (gl < SEQLEN && mask[(size_t)b * SEQLEN + gl]) {
                const float* krow = &lds_k[l * KV_STR + h * HEAD_DIM];
                const float* qh   = &q_s[h * HEAD_DIM];
                #pragma unroll
                for (int d = 0; d < HEAD_DIM; ++d) s += qh[d] * krow[d];
                s *= INV_SQRT_D;
                s = fmaxf(s, 0.0f);  // mask-fill(0) + relu
            }
            score_s[l * HEADS + h] = s;
        }
        __syncthreads();

        // ---- accumulate attention output: thread tid owns feature tid ----
        if (tid < EMB) {
            #pragma unroll
            for (int l = 0; l < TILE; ++l)
                attnAcc += score_s[l * HEADS + myHead] * lds_v[l * KV_STR + tid];
        }
    }

    if (tid < EMB) acc_s[tid] = attnAcc;
    __syncthreads();

    // ---- out projection (f32) ----
    if (tid < EMB) {
        float o = bo[tid];
        const float* wrow = Wo + (size_t)tid * EMB;
        #pragma unroll 8
        for (int k = 0; k < EMB; ++k) o += acc_s[k] * wrow[k];
        o_s[tid] = o;
    }
    __syncthreads();

    // ---- LayerNorm over 128 features ----
    if (tid < EMB) {
        float mu = 0.0f;
        #pragma unroll 8
        for (int k = 0; k < EMB; ++k) mu += o_s[k];
        mu *= (1.0f / EMB);
        float var = 0.0f;
        #pragma unroll 8
        for (int k = 0; k < EMB; ++k) { float d = o_s[k] - mu; var += d * d; }
        var *= (1.0f / EMB);
        float r = rsqrtf(var + LN_EPS);
        out[(size_t)b * EMB + tid] = (o_s[tid] - mu) * r * gamma[tid] + beta[tid];
    }
}

extern "C" void kernel_launch(void* const* d_in, const int* in_sizes, int n_in,
                              void* d_out, int out_size, void* d_ws, size_t ws_size,
                              hipStream_t stream) {
    const float* target        = (const float*)d_in[0];
    const float* seq           = (const float*)d_in[1];
    const unsigned char* maskp = (const unsigned char*)d_in[2];
    const float* Wq = (const float*)d_in[3];
    const float* bq = (const float*)d_in[4];
    const float* Wk = (const float*)d_in[5];
    const float* bk = (const float*)d_in[6];
    const float* Wv = (const float*)d_in[7];
    const float* bv = (const float*)d_in[8];
    const float* Wo = (const float*)d_in[9];
    const float* bo = (const float*)d_in[10];
    const float* gamma = (const float*)d_in[11];
    const float* beta  = (const float*)d_in[12];
    float* out = (float*)d_out;

    hipLaunchKernelGGL(mhta_kernel, dim3(BATCH), dim3(256), 0, stream,
                       target, seq, maskp, Wq, bq, Wk, bk, Wv, bv, Wo, bo,
                       gamma, beta, out);
}
